// NTXentLoss_40364102648130
// MI455X (gfx1250) — compile-verified, matrix operand order checked by reading, not run on hardware
//
#include <hip/hip_runtime.h>
#include <hip/hip_bf16.h>

typedef __attribute__((ext_vector_type(16))) _Float16 v16h;
typedef __attribute__((ext_vector_type(8)))  _Float16 v8h;
typedef __attribute__((ext_vector_type(8)))  float    v8f;

#define NROWS 8192
#define HALF_N 4096
#define D 256
// sqrt(log2(e)/T): WMMA output = sim * log2e / T -> bare v_exp_f32/v_log_f32.
#define FRAG_SCALE 4.5398161f
#define LN2 0.69314718056f

// ---------------------------------------------------------------------------
// Kernel 0: zero the scalar output (harness poisons d_out with 0xAA)
// ---------------------------------------------------------------------------
__global__ void zero_out_kernel(float* out) {
    if (threadIdx.x == 0) out[0] = 0.0f;
}

// ---------------------------------------------------------------------------
// Kernel 1: row-normalize z = concat(z1,z2), scale by sqrt(log2e/T), store f16.
// ---------------------------------------------------------------------------
__global__ __launch_bounds__(256) void normalize_kernel(const float* __restrict__ z1,
                                                        const float* __restrict__ z2,
                                                        _Float16* __restrict__ zn) {
    __shared__ float red[256];
    const int row = blockIdx.x;
    const float* src = (row < HALF_N) ? (z1 + (size_t)row * D)
                                      : (z2 + (size_t)(row - HALF_N) * D);
    const float v = src[threadIdx.x];
    red[threadIdx.x] = v * v;
    __syncthreads();
    #pragma unroll
    for (int s = 128; s > 0; s >>= 1) {
        if (threadIdx.x < s) red[threadIdx.x] += red[threadIdx.x + s];
        __syncthreads();
    }
    const float norm = fmaxf(sqrtf(red[0]), 1e-8f);       // eps clamp per reference
    zn[(size_t)row * D + threadIdx.x] = (_Float16)(v * (FRAG_SCALE / norm));
}

// ---------------------------------------------------------------------------
// Fragment loader for 16-bit 16x32 A/B operand of v_wmma_f32_16x16x32_f16.
// Lanes 0-15: M=0..15, K chunks {0..7, 16..23}; lanes 16-31: K {8..15, 24..31}.
// Gram matrix is symmetric -> B fragments load with the identical pattern.
// ---------------------------------------------------------------------------
__device__ inline v16h load_frag16x32(const _Float16* __restrict__ zn,
                                      int row_base, int kk, int lane) {
    const int m    = lane & 15;
    const int ksel = lane >> 4;
    const _Float16* p = zn + (size_t)(row_base + m) * D + kk * 32 + ksel * 8;
    v8h lo = *(const v8h*)(p);                   // global_load_b128
    v8h hi = *(const v8h*)(p + 16);              // global_load_b128
    v16h r;
    #pragma unroll
    for (int i = 0; i < 8; ++i) { r[i] = lo[i]; r[8 + i] = hi[i]; }
    return r;
}

// Streaming epilogue for one 16x16 C tile (scalar-uniform branch selectors).
__device__ inline void tile_epilogue(int nt, int diag_tile, int pos_tile,
                                     const v8f& c, float* expsum, float* posv,
                                     int col, int ksel) {
    if (nt == diag_tile) {
        #pragma unroll
        for (int r = 0; r < 8; ++r)
            if (col != r + 8 * ksel)
                expsum[r] += __builtin_amdgcn_exp2f(c[r]);           // mask j==i
    } else if (nt == pos_tile) {
        #pragma unroll
        for (int r = 0; r < 8; ++r) {
            expsum[r] += __builtin_amdgcn_exp2f(c[r]);
            if (col == r + 8 * ksel) posv[r] = c[r];                 // positive logit
        }
    } else {
        #pragma unroll
        for (int r = 0; r < 8; ++r)
            expsum[r] += __builtin_amdgcn_exp2f(c[r]);               // bare v_exp_f32
    }
}

// ---------------------------------------------------------------------------
// Kernel 2: fused Gram-GEMM + streaming sum-of-2^s + positive-logit capture.
// One BLOCK = one 32-row tile (two 16-row A panels per wave, 128 VGPRs).
// The block's 4 waves split the 512 column tiles (128 each); each B fragment
// feeds TWO independent WMMAs (two C accumulators) -> 2x arithmetic intensity
// per load and half the total B traffic. Results combined via LDS.
// ---------------------------------------------------------------------------
__global__ __launch_bounds__(128) void ntxent_kernel(const _Float16* __restrict__ zn,
                                                     float* __restrict__ out) {
    __shared__ float lds_e[4][32];
    __shared__ float lds_p[4][32];

    const int lane = threadIdx.x & 31;
    const int wave = __builtin_amdgcn_readfirstlane(threadIdx.x >> 5); // scalar
    const int m0   = blockIdx.x * 32;                                  // uniform
    const int col  = lane & 15;
    const int ksel = lane >> 4;

    // Two A panels: rows m0..m0+15 and m0+16..m0+31 (64 VGPRs each).
    v16h a0[8], a1[8];
    #pragma unroll
    for (int kk = 0; kk < 8; ++kk) {
        a0[kk] = load_frag16x32(zn, m0,      kk, lane);
        a1[kk] = load_frag16x32(zn, m0 + 16, kk, lane);
    }

    float e0[8], p0[8], e1[8], p1[8];
    #pragma unroll
    for (int r = 0; r < 8; ++r) { e0[r] = p0[r] = e1[r] = p1[r] = 0.0f; }

    const int dt0 = 2 * blockIdx.x;             // diag tile for rows m0..m0+15
    const int dt1 = dt0 + 1;                    // diag tile for rows m0+16..m0+31
    const int pt0 = (dt0 + 256) & 511;          // positive tiles
    const int pt1 = (dt1 + 256) & 511;
    const int nt_begin = wave * 128;
    const int nt_end   = nt_begin + 128;

    for (int nt = nt_begin; nt < nt_end; ++nt) {
        const int n0 = nt * 16;

        // Prefetch next B tile (global_prefetch_b8); stream is L2-resident.
        if (nt + 1 < nt_end) {
            const _Float16* np = zn + (size_t)(n0 + 16 + col) * D + ksel * 128;
            __builtin_prefetch(np, 0, 1);
        }

        // Two 16x16 blocks of sim*log2e/T sharing one B stream:
        // per K-step: one load pair -> two independent WMMAs.
        v8f c0 = {}, c1 = {};
        #pragma unroll
        for (int kk = 0; kk < 8; ++kk) {
            v16h b = load_frag16x32(zn, n0, kk, lane);
            c0 = __builtin_amdgcn_wmma_f32_16x16x32_f16(
                     false, a0[kk], false, b, (short)0, c0, false, false);
            c1 = __builtin_amdgcn_wmma_f32_16x16x32_f16(
                     false, a1[kk], false, b, (short)0, c1, false, false);
        }

        tile_epilogue(nt, dt0, pt0, c0, e0, p0, col, ksel);
        tile_epilogue(nt, dt1, pt1, c1, e1, p1, col, ksel);
    }

    // Intra-wave: reduce over the 16 lane-columns within each half-wave.
    #pragma unroll
    for (int r = 0; r < 8; ++r) {
        float a = e0[r], b = p0[r], cc = e1[r], d = p1[r];
        #pragma unroll
        for (int m = 1; m < 16; m <<= 1) {
            a += __shfl_xor(a, m, 32);
            b += __shfl_xor(b, m, 32);
            cc += __shfl_xor(cc, m, 32);
            d += __shfl_xor(d, m, 32);
        }
        e0[r] = a; p0[r] = b; e1[r] = cc; p1[r] = d;
    }

    // Lanes 0 / 16 hold rows +0..7 / +8..15 of each panel -> LDS.
    if (col == 0) {
        #pragma unroll
        for (int r = 0; r < 8; ++r) {
            lds_e[wave][r + 8 * ksel]      = e0[r];
            lds_p[wave][r + 8 * ksel]      = p0[r];
            lds_e[wave][16 + r + 8 * ksel] = e1[r];
            lds_p[wave][16 + r + 8 * ksel] = p1[r];
        }
    }
    __syncthreads();

    // Cross-wave combine: 32 threads, one per row of the 32-row tile.
    // loss_row = ln2 * (log2(sum 2^s) - s_pos); mean over 2N rows.
    if (threadIdx.x < 32) {
        float e = 0.0f, p = 0.0f;
        #pragma unroll
        for (int w = 0; w < 4; ++w) { e += lds_e[w][threadIdx.x]; p += lds_p[w][threadIdx.x]; }
        float part = __builtin_amdgcn_logf(e) - p;                   // bare v_log_f32
        #pragma unroll
        for (int m = 1; m < 32; m <<= 1) part += __shfl_xor(part, m, 32);
        if (threadIdx.x == 0) atomicAdd(out, part * (LN2 / (float)NROWS));
    }
}

// ---------------------------------------------------------------------------
extern "C" void kernel_launch(void* const* d_in, const int* in_sizes, int n_in,
                              void* d_out, int out_size, void* d_ws, size_t ws_size,
                              hipStream_t stream) {
    const float* z1 = (const float*)d_in[0];
    const float* z2 = (const float*)d_in[1];
    float* out = (float*)d_out;
    _Float16* zn = (_Float16*)d_ws;   // 8192*256*2 = 4 MB scratch

    zero_out_kernel<<<1, 32, 0, stream>>>(out);
    normalize_kernel<<<NROWS, 256, 0, stream>>>(z1, z2, zn);
    ntxent_kernel<<<256, 128, 0, stream>>>(zn, out);
}
